// ColorHead_26233660244121
// MI455X (gfx1250) — matrix operand order repaired and checked by reading, not compiled, not run on hardware
//
#include <hip/hip_runtime.h>

// ---------------- problem constants ----------------
#define BATCH 8
#define CIN   3
#define HIN   512
#define WIN   512
#define HO    504
#define WO    504
#define PPLANE (HO*WO)            // 254016  (divisible by 32)
#define NPIX   (BATCH*PPLANE)     // 2032128 (divisible by 32)
#define CH     32
#define NCHUNK (NPIX/32)          // 63504 wave chunks of 32 pixels
#define NBLK   512
#define BLKT   256
#define WAVES_PER_BLK (BLKT/32)   // 8 waves (wave32)
#define TOTWAVES (NBLK*WAVES_PER_BLK)
#define BN_EPS 1e-5f
#define SLOPE  0.01f

typedef __attribute__((ext_vector_type(16))) _Float16 v16h;
typedef __attribute__((ext_vector_type(8)))  float    v8f;

// =====================================================================
// Kernel 1: fused ColorDownsample + ModePool2d(11, stride 1, pad 1)
//   bin = clip(rint(x * 255/16), 0, 16)   (double-round collapses)
//   Interior tiles: async global->LDS DMA of the float tile (ASYNCcnt),
//   border tiles: bounds-checked scalar path.
//   Per-thread 17-bin byte histogram in LDS, first-argmax scan.
// =====================================================================
#define TX 24
#define TY 8
#define TILW (TX+10)   // 34
#define TILH (TY+10)   // 18
#define TILN (TILW*TILH) // 612

__global__ __launch_bounds__(TX*TY)
void mode_kernel(const float* __restrict__ x, unsigned char* __restrict__ mb) {
  __shared__ float tilef[TILN];
  __shared__ unsigned char tile[TILN];
  __shared__ unsigned char hist[TX*TY*32];
  const int tid   = threadIdx.x;
  const int plane = blockIdx.z;                 // b*3 + c
  const int x0 = blockIdx.x*TX, y0 = blockIdx.y*TY;
  const float* xp = x + (size_t)plane*HIN*WIN;

  // block-uniform interior test: window x in [x0-1, x0+TX+9], y likewise
  const bool interior = (x0 >= 1) && (x0 + TX + 9 < WIN) &&
                        (y0 >= 1) && (y0 + TY + 9 < HIN);
  if (interior) {
    // async DMA the 34x18 float tile into LDS (GVS: SGPR base + VGPR offset)
    const unsigned long long sbase = (unsigned long long)(uintptr_t)xp;
    for (int i = tid; i < TILN; i += TX*TY) {
      int ty = i / TILW, tx = i - ty*TILW;
      unsigned goff  = (unsigned)(((y0 + ty - 1)*WIN + (x0 + tx - 1)) * 4);
      unsigned laddr = (unsigned)(uintptr_t)&tilef[i];   // LDS byte address
      asm volatile("global_load_async_to_lds_b32 %0, %1, %2"
                   :: "v"(laddr), "v"(goff), "s"(sbase) : "memory");
    }
    asm volatile("s_wait_asynccnt 0x0" ::: "memory");
    for (int i = tid; i < TILN; i += TX*TY) {
      float f = rintf(tilef[i]*15.9375f);                // RNE, like jnp.round
      f = fminf(fmaxf(f, 0.f), 16.f);
      tile[i] = (unsigned char)f;
    }
  } else {
    for (int i = tid; i < TILN; i += TX*TY) {
      int ty = i / TILW, tx = i - ty*TILW;
      int gy = y0 + ty - 1, gx = x0 + tx - 1;            // pad=1 border -> bin 0
      unsigned char v = 0;
      if (gy >= 0 && gy < HIN && gx >= 0 && gx < WIN) {
        float f = rintf(xp[gy*WIN + gx]*15.9375f);
        f = fminf(fmaxf(f, 0.f), 16.f);
        v = (unsigned char)f;
      }
      tile[i] = v;
    }
  }
  __syncthreads();

  unsigned int* hz = (unsigned int*)&hist[tid*32];
  #pragma unroll
  for (int i = 0; i < 8; ++i) hz[i] = 0u;

  const int tyl = tid / TX, txl = tid - tyl*TX;
  unsigned char* h = &hist[tid*32];
  #pragma unroll
  for (int ky = 0; ky < 11; ++ky) {
    #pragma unroll
    for (int kx = 0; kx < 11; ++kx) {
      unsigned char v = tile[(tyl+ky)*TILW + txl + kx];
      h[v] = (unsigned char)(h[v] + 1);
    }
  }
  int best = 0, bc = h[0];
  #pragma unroll
  for (int b = 1; b < 17; ++b) { int c = h[b]; if (c > bc) { bc = c; best = b; } }
  mb[(size_t)plane*PPLANE + (y0+tyl)*WO + (x0+txl)] = (unsigned char)best;
}

// =====================================================================
// Kernel 2: per-channel sum / sumsq of y1 = fold(conv1,dw1)(mode)
//   weights rounded through f16 so stats match the WMMA conv path.
// =====================================================================
__global__ __launch_bounds__(BLKT)
void stats1_kernel(const unsigned char* __restrict__ mb,
                   const float* __restrict__ w1, const float* __restrict__ b1,
                   const float* __restrict__ wd1, const float* __restrict__ bd1,
                   float* __restrict__ part) {
  __shared__ float w1p[CH*3], c1[CH], accum[2*CH];
  const int tid = threadIdx.x;
  if (tid < CH*3) w1p[tid] = (float)(_Float16)(wd1[tid/3]*w1[tid]);
  if (tid < CH)   c1[tid]  = wd1[tid]*b1[tid] + bd1[tid];
  if (tid < 2*CH) accum[tid] = 0.f;
  __syncthreads();

  float sA[CH], sQ[CH];
  #pragma unroll
  for (int c = 0; c < CH; ++c) { sA[c] = 0.f; sQ[c] = 0.f; }

  const int lane = tid & 31;
  const int wave = blockIdx.x*WAVES_PER_BLK + (tid>>5);
  for (int chunk = wave; chunk < NCHUNK; chunk += TOTWAVES) {
    int qb = chunk*32;
    int b  = qb / PPLANE;
    int p  = qb - b*PPLANE + lane;
    size_t base = (size_t)b*3*PPLANE + p;
    float m0 = mb[base           ]*0.0625f;
    float m1 = mb[base +   PPLANE]*0.0625f;
    float m2 = mb[base + 2*PPLANE]*0.0625f;
    #pragma unroll
    for (int c = 0; c < CH; ++c) {
      float f = fmaf(w1p[3*c], m0, fmaf(w1p[3*c+1], m1, fmaf(w1p[3*c+2], m2, c1[c])));
      sA[c] += f; sQ[c] += f*f;
    }
  }
  #pragma unroll
  for (int c = 0; c < CH; ++c) { atomicAdd(&accum[c], sA[c]); atomicAdd(&accum[CH+c], sQ[c]); }
  __syncthreads();
  if (tid < 2*CH) part[blockIdx.x*2*CH + tid] = accum[tid];
}

// deterministic fixed-order final reduction of per-block partials
__global__ void reduce_kernel(const float* __restrict__ part, float* __restrict__ stats) {
  int j = threadIdx.x;     // 64 threads
  float s = 0.f;
  for (int i = 0; i < NBLK; ++i) s += part[i*64 + j];
  stats[j] = s;
}

// =====================================================================
// Kernels 3/4: both 1x1 convs on the matrix pipe.
//   conv1: A = folded 32x3 weights zero-padded to K=32, B = mode pixels.
//   D-tiles get bias+BN1+leaky in registers, packed f16 into LDS [px][ch].
//   conv2: A = folded 32x32 weights, B gathered from LDS (same-wave DS
//   in-order, fenced with s_wait_dscnt 0).
//   FINAL=false: accumulate BN2 stats.  FINAL=true: BN2+leaky -> d_out.
// =====================================================================
template<bool FINAL>
__global__ __launch_bounds__(BLKT)
void conv_kernel(const unsigned char* __restrict__ mb,
                 const float* __restrict__ w1,  const float* __restrict__ b1,
                 const float* __restrict__ wd1, const float* __restrict__ bd1,
                 const float* __restrict__ g1,  const float* __restrict__ be1,
                 const float* __restrict__ w2,  const float* __restrict__ b2,
                 const float* __restrict__ wd2, const float* __restrict__ bd2,
                 const float* __restrict__ g2,  const float* __restrict__ be2,
                 const float* __restrict__ stats1, const float* __restrict__ stats2,
                 float* __restrict__ part2, float* __restrict__ out) {
  __shared__ float c1[CH], s1[CH], t1[CH], c2[CH], s2[CH], t2[CH];
  __shared__ float accum[2*CH];
  __shared__ unsigned int h1buf[WAVES_PER_BLK*32*16];  // per wave: 32 px x 32 ch f16

  const int tid = threadIdx.x;
  if (tid < CH) {
    c1[tid] = wd1[tid]*b1[tid] + bd1[tid];
    float mean = stats1[tid]*(1.f/NPIX);
    float var  = stats1[CH+tid]*(1.f/NPIX) - mean*mean;   // biased var, like jnp.var
    float inv  = rsqrtf(var + BN_EPS);
    s1[tid] = g1[tid]*inv;
    t1[tid] = be1[tid] - mean*s1[tid];
    c2[tid] = wd2[tid]*b2[tid] + bd2[tid];
    if (FINAL) {
      float mean2 = stats2[tid]*(1.f/NPIX);
      float var2  = stats2[CH+tid]*(1.f/NPIX) - mean2*mean2;
      float inv2  = rsqrtf(var2 + BN_EPS);
      s2[tid] = g2[tid]*inv2;
      t2[tid] = be2[tid] - mean2*s2[tid];
    }
  }
  if (tid < 2*CH) accum[tid] = 0.f;
  __syncthreads();

  const int lane = tid & 31;
  const int hi   = lane >> 4;     // half-wave group
  const int mrow = lane & 15;
  const int wv   = tid >> 5;

  // conv1 A matrices (16x32 f16 A layout; only K=0..2 nonzero, lanes hi==0)
  v16h a1[2];
  #pragma unroll
  for (int o = 0; o < 2; ++o) {
    int row = o*16 + mrow;
    v16h a;
    #pragma unroll
    for (int j = 0; j < 16; ++j) a[j] = (_Float16)0.f;
    if (hi == 0) {
      float sc = wd1[row];
      #pragma unroll
      for (int k = 0; k < 3; ++k) a[k] = (_Float16)(sc * w1[row*3 + k]);
    }
    a1[o] = a;
  }

  // conv2 A matrices: folded weights, contiguous k order (k = 16*hi + j)
  v16h aW[2];
  #pragma unroll
  for (int o = 0; o < 2; ++o) {
    int row = o*16 + mrow;
    float sc = wd2[row];
    v16h a;
    #pragma unroll
    for (int j = 0; j < 8; ++j) a[j]   = (_Float16)(sc * w2[row*CH + hi*16 +     j]);
    #pragma unroll
    for (int j = 0; j < 8; ++j) a[8+j] = (_Float16)(sc * w2[row*CH + hi*16 + 8 + j]);
    aW[o] = a;
  }

  float sA[16], sQ[16];
  #pragma unroll
  for (int i = 0; i < 16; ++i) { sA[i] = 0.f; sQ[i] = 0.f; }

  const int wave = blockIdx.x*WAVES_PER_BLK + wv;
  for (int chunk = wave; chunk < NCHUNK; chunk += TOTWAVES) {
    int qb = chunk*32;
    int b  = qb / PPLANE;
    int p0 = qb - b*PPLANE;                      // chunk stays inside one image
    size_t pbase = (size_t)b*3*PPLANE + p0;

    // prefetch next chunk's mode bytes
    int nc = chunk + TOTWAVES;
    if (nc < NCHUNK) {
      int nb = (nc*32) / PPLANE;
      __builtin_prefetch(&mb[(size_t)nb*3*PPLANE + (nc*32 - nb*PPLANE) + lane], 0, 1);
    }

    // conv1 B matrices: lanes 0-15 carry pixel t*16+mrow, K=0..2; rest zero
    v16h b1m[2];
    #pragma unroll
    for (int t = 0; t < 2; ++t) {
      v16h bb;
      #pragma unroll
      for (int j = 0; j < 16; ++j) bb[j] = (_Float16)0.f;
      b1m[t] = bb;
    }
    if (hi == 0) {
      #pragma unroll
      for (int t = 0; t < 2; ++t) {
        size_t pp = pbase + t*16 + mrow;
        v16h bb = b1m[t];
        bb[0] = (_Float16)(mb[pp           ]*0.0625f);   // exact in f16
        bb[1] = (_Float16)(mb[pp +   PPLANE]*0.0625f);
        bb[2] = (_Float16)(mb[pp + 2*PPLANE]*0.0625f);
        b1m[t] = bb;
      }
    }

    // conv1 on the matrix pipe
    v8f d1[2][2];
    #pragma unroll
    for (int t = 0; t < 2; ++t)
      #pragma unroll
      for (int o = 0; o < 2; ++o) {
        v8f cz = {};
        d1[o][t] = __builtin_amdgcn_wmma_f32_16x16x32_f16(
            false, a1[o], false, b1m[t], (short)0, cz, false, false);
      }

    // bias + BN1 + leaky, pack f16, stage into LDS [px][ch]
    #pragma unroll
    for (int t = 0; t < 2; ++t) {
      int px = t*16 + mrow;
      #pragma unroll
      for (int o = 0; o < 2; ++o) {
        unsigned int hpk[4];
        unsigned int lo16 = 0;
        #pragma unroll
        for (int r = 0; r < 8; ++r) {
          int ch = o*16 + hi*8 + r;
          float f = d1[o][t][r] + c1[ch];
          f = s1[ch]*f + t1[ch];
          f = (f >= 0.f) ? f : SLOPE*f;              // LeakyReLU
          _Float16 hh = (_Float16)f;
          unsigned short u; __builtin_memcpy(&u, &hh, 2);
          if ((r & 1) == 0) lo16 = (unsigned int)u;
          else             hpk[r>>1] = lo16 | ((unsigned int)u << 16);
        }
        *((uint4*)&h1buf[wv*512 + px*16 + o*8 + hi*4]) = *((uint4*)hpk);
      }
    }
    asm volatile("s_wait_dscnt 0" ::: "memory");       // same-wave DS is in-order

    // conv2: gather B (32ch x 16px) from LDS, 1 WMMA covers K=32
    v8f d[2][2];
    #pragma unroll
    for (int t = 0; t < 2; ++t) {
      union { v16h v; uint4 q[2]; } B;                 // lane<16 -> K0..15, else K16..31
      const unsigned int* src = &h1buf[wv*512 + (t*16 + mrow)*16 + hi*8];
      B.q[0] = ((const uint4*)src)[0];
      B.q[1] = ((const uint4*)src)[1];
      #pragma unroll
      for (int o = 0; o < 2; ++o) {
        v8f cz = {};
        d[o][t] = __builtin_amdgcn_wmma_f32_16x16x32_f16(
            false, aW[o], false, B.v, (short)0, cz, false, false);
      }
    }
    #pragma unroll
    for (int o = 0; o < 2; ++o) {
      #pragma unroll
      for (int t = 0; t < 2; ++t) {
        #pragma unroll
        for (int r = 0; r < 8; ++r) {
          int ch = o*16 + hi*8 + r;                    // D: M = r + 8*hi, N = lane&15
          float v = d[o][t][r] + c2[ch];
          if (FINAL) {
            float w = s2[ch]*v + t2[ch];
            w = (w >= 0.f) ? w : SLOPE*w;
            out[((size_t)b*CH + ch)*PPLANE + p0 + t*16 + mrow] = w;
          } else {
            sA[o*8+r] += v; sQ[o*8+r] += v*v;
          }
        }
      }
    }
  }

  if (!FINAL) {
    #pragma unroll
    for (int o = 0; o < 2; ++o)
      #pragma unroll
      for (int r = 0; r < 8; ++r) {
        int ch = o*16 + hi*8 + r;
        atomicAdd(&accum[ch],    sA[o*8+r]);
        atomicAdd(&accum[CH+ch], sQ[o*8+r]);
      }
    __syncthreads();
    if (tid < 2*CH) part2[blockIdx.x*2*CH + tid] = accum[tid];
  }
}

// =====================================================================
extern "C" void kernel_launch(void* const* d_in, const int* in_sizes, int n_in,
                              void* d_out, int out_size, void* d_ws, size_t ws_size,
                              hipStream_t stream) {
  const float* x   = (const float*)d_in[0];
  const float* w1  = (const float*)d_in[1];
  const float* b1  = (const float*)d_in[2];
  const float* wd1 = (const float*)d_in[3];
  const float* bd1 = (const float*)d_in[4];
  const float* g1  = (const float*)d_in[5];
  const float* be1 = (const float*)d_in[6];
  const float* w2  = (const float*)d_in[7];
  const float* b2  = (const float*)d_in[8];
  const float* wd2 = (const float*)d_in[9];
  const float* bd2 = (const float*)d_in[10];
  const float* g2  = (const float*)d_in[11];
  const float* be2 = (const float*)d_in[12];
  float* out = (float*)d_out;

  // workspace layout: mode bins (u8) | part1 | part2 | stats1 | stats2
  unsigned char* mb = (unsigned char*)d_ws;
  size_t mb_bytes = (size_t)BATCH*CIN*PPLANE;            // 6,096,384 (256-aligned)
  float* part1  = (float*)((char*)d_ws + mb_bytes);
  float* part2  = part1 + (size_t)NBLK*64;
  float* stats1 = part2 + (size_t)NBLK*64;
  float* stats2 = stats1 + 64;

  dim3 mgrid(WO/TX, HO/TY, BATCH*CIN);                   // 21 x 63 x 24
  mode_kernel<<<mgrid, TX*TY, 0, stream>>>(x, mb);

  stats1_kernel<<<NBLK, BLKT, 0, stream>>>(mb, w1, b1, wd1, bd1, part1);
  reduce_kernel<<<1, 64, 0, stream>>>(part1, stats1);

  conv_kernel<false><<<NBLK, BLKT, 0, stream>>>(mb, w1, b1, wd1, bd1, g1, be1,
                                                w2, b2, wd2, bd2, g2, be2,
                                                stats1, stats2, part2, out);
  reduce_kernel<<<1, 64, 0, stream>>>(part2, stats2);

  conv_kernel<true><<<NBLK, BLKT, 0, stream>>>(mb, w1, b1, wd1, bd1, g1, be1,
                                               w2, b2, wd2, bd2, g2, be2,
                                               stats1, stats2, part2, out);
}
